// PointTransformerLayer_91268055040155
// MI455X (gfx1250) — matrix operand order repaired and checked by reading, not compiled, not run on hardware
//
#include <hip/hip_runtime.h>

// ---------------------------------------------------------------------------
// PointTransformerLayer for MI455X (gfx1250, wave32, WMMA + TDM)
//   N = 50000 points, C = 64 channels, K = 16 neighbors
// Strategy:
//   kernel 0 (prep): transpose+convert all 64x64 weights to f16 once into ws.
//   kernel 1: q/k/v = feat @ W + b  (f16 WMMA, f32 accum), results to ws (f32)
//             -> k/v (12.8 MB each) stay resident in the 192 MB L2 so the
//                per-point gathers below are L2 hits.
//   kernel 2: one wave per point. 16 neighbors == 16 rows of the WMMA
//             A-matrix. p = relu(LN(rel@wp+bp)) in VALU (K-dim is only 3),
//             then attn@wg1 and h@wg2 as 16x16x32 f16 WMMA (8 each),
//             LN across channels via lane-group shuffles, softmax across the
//             16 neighbor rows via shfl_xor(16), weighted gv+p sum.
//   kernel 3: out2 @ wo + bo + residual (f16 WMMA).
//   Weight tiles are staged into LDS with the Tensor Data Mover
//   (tensor_load_to_lds + s_wait_tensorcnt) instead of per-thread copies.
// ---------------------------------------------------------------------------

typedef _Float16 v16h __attribute__((ext_vector_type(16)));
typedef _Float16 v8h  __attribute__((ext_vector_type(8)));
typedef float    v8f  __attribute__((ext_vector_type(8)));
typedef unsigned int v4u __attribute__((ext_vector_type(4)));
typedef int v8i __attribute__((ext_vector_type(8)));
typedef int v4i __attribute__((ext_vector_type(4)));

#define LN_EPS 1e-5f

#if __has_builtin(__builtin_amdgcn_tensor_load_to_lds) && \
    __has_builtin(__builtin_amdgcn_s_wait_tensorcnt)
#define HAVE_TDM 1
#else
#define HAVE_TDM 0
#endif

static __device__ __forceinline__ v8f wmma16x16x32(v16h a, v16h b, v8f c) {
  // D = A(16x32 f16) x B(32x16 f16) + C(16x16 f32)
  return __builtin_amdgcn_wmma_f32_16x16x32_f16(false, a, false, b, (short)0, c,
                                                false, false);
}

static __device__ __forceinline__ v16h cat8(v8h lo, v8h hi) {
  return __builtin_shufflevector(lo, hi, 0, 1, 2, 3, 4, 5, 6, 7, 8, 9, 10, 11,
                                 12, 13, 14, 15);
}

// A-matrix fragment: rowbase = &tile[m*64] (row-major f16 16x64 tile in LDS).
// ISA 16-bit A 16x32 layout: lane(m,half) holds K = kb*32 + half*8 + {0..7}
// in h[0..7] and K = kb*32 + 16 + half*8 + {0..7} in h[8..15].
static __device__ __forceinline__ v16h load_a(const _Float16* rowbase, int kb,
                                              int half) {
  const _Float16* p = rowbase + kb * 32 + half * 8;
  v8h lo = *(const v8h*)p;
  v8h hi = *(const v8h*)(p + 16);
  return cat8(lo, hi);
}

// B-matrix fragment from transposed weight WT[n*64 + k] (f16 in LDS).
// ISA 16-bit B 32x16 layout: lane(n=m,half) holds K = kb*32 + half*8 + {0..7}
// in h[0..7] and K = kb*32 + 16 + half*8 + {0..7} in h[8..15].
static __device__ __forceinline__ v16h load_b(const _Float16* WT, int ncol,
                                              int kb, int half) {
  const _Float16* p = WT + ncol * 64 + kb * 32 + half * 8;
  v8h lo = *(const v8h*)p;
  v8h hi = *(const v8h*)(p + 16);
  return cat8(lo, hi);
}

// Wave-internal LDS producer->consumer fence (cross-lane via LDS, no block
// barrier needed: all traffic is within one wave32).
static __device__ __forceinline__ void wave_lds_fence() {
#if __has_builtin(__builtin_amdgcn_wave_barrier)
  __builtin_amdgcn_wave_barrier();
#endif
  asm volatile("s_wait_dscnt 0x0" ::: "memory");
#if __has_builtin(__builtin_amdgcn_wave_barrier)
  __builtin_amdgcn_wave_barrier();
#endif
}

// Butterfly sum across a 16-lane group (stays within half).
static __device__ __forceinline__ float sum16(float x) {
  x += __shfl_xor(x, 1, 32);
  x += __shfl_xor(x, 2, 32);
  x += __shfl_xor(x, 4, 32);
  x += __shfl_xor(x, 8, 32);
  return x;
}

// ---------------------------------------------------------------------------
// TDM: DMA a contiguous run of e8 8-byte elements from global f16 (already
// transposed weights in ws) into LDS. Descriptor per CDNA5 ISA ch.8:
//   group0: count=1 | lds_addr | global_addr[56:0] | type=2
//   group1: data_size=3 (8B), tensor_dim0=tile_dim0=e8, dims1..=1, stride0=e8
// Issued by one wave (EXEC is ignored by TDM; guard selects a single wave),
// completion via s_wait_tensorcnt 0 then the workgroup barrier publishes it.
// This toolchain declares the 6-arg builtin:
//   (uint32x4 g0, int32x8 g1, int32x4, int32x4, int32x8, i32 cpol)
// ---------------------------------------------------------------------------
static __device__ __forceinline__ void tdm_load_1d(const _Float16* gsrc,
                                                   _Float16* lds_dst,
                                                   unsigned e8) {
#if HAVE_TDM
  const unsigned long long ga = (unsigned long long)(uintptr_t)gsrc;
  const unsigned lds = (unsigned)(uintptr_t)lds_dst;  // addr[31:0] == LDS byte
  v4u g0;
  g0.x = 1u;                                              // count=1, no gather
  g0.y = lds;                                             // lds_addr
  g0.z = (unsigned)ga;                                    // global_addr[31:0]
  g0.w = (unsigned)((ga >> 32) & 0x01FFFFFFu) | (2u << 30);  // [56:32] | type=2
  v8i g1;
  g1[0] = 0x00030000;                       // workgroup_mask=0, data_size=8B
  g1[1] = (int)((e8 & 0xFFFFu) << 16);      // tensor_dim0[15:0]
  g1[2] = (int)((e8 >> 16) | (1u << 16));   // tensor_dim0[31:16], tensor_dim1=1
  g1[3] = (int)(e8 << 16);                  // tile_dim0 = e8 (e8 <= 0xFFFF)
  g1[4] = 1;                                // tile_dim1=1, tile_dim2=0
  g1[5] = (int)e8;                          // tensor_dim0_stride[31:0]
  g1[6] = 0;                                // stride0 hi, stride1 lo
  g1[7] = 0;
  const v4i z4 = {0, 0, 0, 0};
  const v8i z8 = {0, 0, 0, 0, 0, 0, 0, 0};
  __builtin_amdgcn_tensor_load_to_lds(g0, g1, z4, z4, z8, 0);
#endif
}

// Stage weights into LDS: TDM path (one wave) or plain copy fallback.
// Caller must __syncthreads() afterwards.
static __device__ __forceinline__ void stage_weights(const _Float16* gsrc,
                                                     _Float16* lds_dst,
                                                     unsigned halves, int t,
                                                     int nthreads) {
#if HAVE_TDM
  if (t < 32) {
    tdm_load_1d(gsrc, lds_dst, halves >> 2);
    __builtin_amdgcn_s_wait_tensorcnt(0);
  }
#else
  const unsigned e8 = halves >> 2;
  unsigned long long* d = (unsigned long long*)lds_dst;
  const unsigned long long* s = (const unsigned long long*)gsrc;
  for (unsigned i = t; i < e8; i += nthreads) d[i] = s[i];
#endif
}

// ---------------------------------------------------------------------------
// Kernel 0: transpose + f16-convert the six 64x64 weights into ws.
// Order: wq, wk, wv, wg1, wg2, wo  (each stored as WT[n*64+k] = W[k*64+n]).
// ---------------------------------------------------------------------------
__global__ __launch_bounds__(256) void prep_weights(
    const float* __restrict__ wq, const float* __restrict__ wk,
    const float* __restrict__ wv, const float* __restrict__ wg1,
    const float* __restrict__ wg2, const float* __restrict__ wo,
    _Float16* __restrict__ dst) {
  const int e = blockIdx.x * 256 + threadIdx.x;
  if (e >= 6 * 4096) return;
  const int which = e >> 12;
  const int r = e & 4095;
  const int n = r >> 6, k = r & 63;
  const float* W = which == 0   ? wq
                   : which == 1 ? wk
                   : which == 2 ? wv
                   : which == 3 ? wg1
                   : which == 4 ? wg2
                                : wo;
  dst[e] = (_Float16)W[k * 64 + n];
}

// ---------------------------------------------------------------------------
// Kernel 1: q/k/v projections.  1 wave = 16 rows; block = 4 waves = 64 rows.
// ---------------------------------------------------------------------------
__global__ __launch_bounds__(128) void qkv_kernel(
    const float* __restrict__ feat, const _Float16* __restrict__ wT3,
    const float* __restrict__ bq, const float* __restrict__ bk,
    const float* __restrict__ bv, float* __restrict__ qout,
    float* __restrict__ kout, float* __restrict__ vout, int N) {
  __shared__ __attribute__((aligned(32))) _Float16 WT[3 * 64 * 64];
  __shared__ float BIAS[3][64];
  __shared__ __attribute__((aligned(32))) _Float16 AT[4][16 * 64];

  const int t = threadIdx.x;
  stage_weights(wT3, WT, 3 * 4096, t, 128);
  if (t < 64) {
    BIAS[0][t] = bq[t];
    BIAS[1][t] = bk[t];
    BIAS[2][t] = bv[t];
  }
  __syncthreads();

  const int wid = t >> 5;
  const int lane = t & 31;
  const int m = lane & 15, half = lane >> 4;
  const long row0 = ((long)blockIdx.x * 4 + wid) * 16;
  if (row0 >= N) return;

  // Stage this wave's 16x64 feature tile as f16 (row-major).
  {
    const float* src = feat + (row0 + m) * 64 + half * 32;
    _Float16* dst = &AT[wid][m * 64 + half * 32];
#pragma unroll
    for (int j = 0; j < 32; j += 4) {
      const float4 f = *(const float4*)(src + j);
      dst[j + 0] = (_Float16)f.x;
      dst[j + 1] = (_Float16)f.y;
      dst[j + 2] = (_Float16)f.z;
      dst[j + 3] = (_Float16)f.w;
    }
  }
  wave_lds_fence();

  const v16h A0 = load_a(&AT[wid][m * 64], 0, half);
  const v16h A1 = load_a(&AT[wid][m * 64], 1, half);

  for (int wsel = 0; wsel < 3; ++wsel) {
    float* dstbuf = wsel == 0 ? qout : (wsel == 1 ? kout : vout);
    const _Float16* W = WT + wsel * 4096;
#pragma unroll
    for (int nt = 0; nt < 4; ++nt) {
      const int ncol = nt * 16 + m;
      v8f acc = {};
      acc = wmma16x16x32(A0, load_b(W, ncol, 0, half), acc);
      acc = wmma16x16x32(A1, load_b(W, ncol, 1, half), acc);
      const float bias = BIAS[wsel][ncol];
      float* dst = dstbuf + (row0 + half * 8) * 64 + ncol;
#pragma unroll
      for (int r = 0; r < 8; ++r) dst[r * 64] = acc[r] + bias;
    }
  }
}

// ---------------------------------------------------------------------------
// Kernel 2: fused per-point attention.  1 wave = 1 point (16 neighbor rows).
// ---------------------------------------------------------------------------
__global__ __launch_bounds__(128) void attn_kernel(
    const float* __restrict__ points, const int* __restrict__ nidx,
    const float* __restrict__ qbuf, const float* __restrict__ kbuf,
    const float* __restrict__ vbuf, const _Float16* __restrict__ wT12,
    const float* __restrict__ wp, const float* __restrict__ bp,
    const float* __restrict__ gp, const float* __restrict__ betap,
    const float* __restrict__ bg1, const float* __restrict__ gg,
    const float* __restrict__ betag, const float* __restrict__ bg2,
    float* __restrict__ out2, int N) {
  __shared__ __attribute__((aligned(32))) _Float16 W12T[2 * 64 * 64];
  __shared__ float WP[3 * 64];
  __shared__ float S_bp[64], S_gp[64], S_betap[64];
  __shared__ float S_bg1[64], S_gg[64], S_betag[64], S_bg2[64];
  __shared__ int IDX[4][16];
  __shared__ __attribute__((aligned(16))) float REL[4][16][4];
  __shared__ __attribute__((aligned(16))) float QN[4][64];
  __shared__ __attribute__((aligned(16))) float PBUF[4][16 * 64];
  __shared__ __attribute__((aligned(32))) _Float16 ATT[4][16 * 64];
  __shared__ __attribute__((aligned(32))) _Float16 HBUF[4][16 * 64];

  const int t = threadIdx.x;
  stage_weights(wT12, W12T, 2 * 4096, t, 128);
  for (int e = t; e < 192; e += 128) WP[e] = wp[e];
  if (t < 64) {
    S_bp[t] = bp[t];
    S_gp[t] = gp[t];
    S_betap[t] = betap[t];
    S_bg1[t] = bg1[t];
    S_gg[t] = gg[t];
    S_betag[t] = betag[t];
    S_bg2[t] = bg2[t];
  }
  __syncthreads();

  const _Float16* W1T = W12T;
  const _Float16* W2T = W12T + 4096;
  const int wid = t >> 5, lane = t & 31;
  const int m = lane & 15, half = lane >> 4;
  const long n = (long)blockIdx.x * 4 + wid;
  if (n >= N) return;

  // ---- stage neighbor ids, relative coords, q-row ----
  if (half == 0) {
    const int id = nidx[n * 16 + m];
    IDX[wid][m] = id;
    REL[wid][m][0] = points[(long)id * 3 + 0] - points[n * 3 + 0];
    REL[wid][m][1] = points[(long)id * 3 + 1] - points[n * 3 + 1];
    REL[wid][m][2] = points[(long)id * 3 + 2] - points[n * 3 + 2];
  }
  QN[wid][lane] = qbuf[n * 64 + lane];
  QN[wid][lane + 32] = qbuf[n * 64 + 32 + lane];
  wave_lds_fence();

  const int cbase = half * 32;  // lane handles row m, columns [cbase, cbase+32)

  // ---- p = relu(LN(rel @ wp + bp)) ----
  {
    const float r0 = REL[wid][m][0], r1 = REL[wid][m][1], r2 = REL[wid][m][2];
    float pv[32];
    float s1 = 0.f, s2 = 0.f;
#pragma unroll
    for (int j = 0; j < 32; ++j) {
      const int c = cbase + j;
      const float x =
          fmaf(r0, WP[c], fmaf(r1, WP[64 + c], fmaf(r2, WP[128 + c], S_bp[c])));
      pv[j] = x;
      s1 += x;
      s2 += x * x;
    }
    s1 += __shfl_xor(s1, 16, 32);  // other 32 channels live in lane^16
    s2 += __shfl_xor(s2, 16, 32);
    const float mu = s1 * (1.f / 64.f);
    const float inv = rsqrtf(s2 * (1.f / 64.f) - mu * mu + LN_EPS);
#pragma unroll
    for (int j = 0; j < 32; ++j) {
      const int c = cbase + j;
      const float x = (pv[j] - mu) * inv * S_gp[c] + S_betap[c];
      PBUF[wid][m * 64 + c] = fmaxf(x, 0.f);
    }
  }
  wave_lds_fence();

  // ---- attn_in = q - gk + p  -> f16 A-tile in LDS (gk gather hits L2) ----
  {
    const long idm = IDX[wid][m];
    const float* krow = kbuf + idm * 64;
#pragma unroll
    for (int j = 0; j < 32; j += 4) {
      const int c = cbase + j;
      const float4 kk = *(const float4*)(krow + c);
      const float4 qq = *(const float4*)(&QN[wid][c]);
      const float4 pp = *(const float4*)(&PBUF[wid][m * 64 + c]);
      ATT[wid][m * 64 + c + 0] = (_Float16)(qq.x - kk.x + pp.x);
      ATT[wid][m * 64 + c + 1] = (_Float16)(qq.y - kk.y + pp.y);
      ATT[wid][m * 64 + c + 2] = (_Float16)(qq.z - kk.z + pp.z);
      ATT[wid][m * 64 + c + 3] = (_Float16)(qq.w - kk.w + pp.w);
    }
  }
  wave_lds_fence();

  // ---- GEMM1: attn(16x64) @ wg1(64x64) + bg1, LN over channels, relu ----
  {
    const v16h A0 = load_a(&ATT[wid][m * 64], 0, half);
    const v16h A1 = load_a(&ATT[wid][m * 64], 1, half);
    v8f acc[4];
#pragma unroll
    for (int nt = 0; nt < 4; ++nt) {
      const int ncol = nt * 16 + m;
      v8f a = {};
      a = wmma16x16x32(A0, load_b(W1T, ncol, 0, half), a);
      a = wmma16x16x32(A1, load_b(W1T, ncol, 1, half), a);
      const float b = S_bg1[ncol];
#pragma unroll
      for (int r = 0; r < 8; ++r) a[r] += b;
      acc[nt] = a;
    }
#pragma unroll
    for (int r = 0; r < 8; ++r) {
      const float x0 = acc[0][r], x1 = acc[1][r], x2 = acc[2][r],
                  x3 = acc[3][r];
      float s1 = sum16(x0 + x1 + x2 + x3);
      float s2 = sum16(x0 * x0 + x1 * x1 + x2 * x2 + x3 * x3);
      const float mu = s1 * (1.f / 64.f);
      const float inv = rsqrtf(s2 * (1.f / 64.f) - mu * mu + LN_EPS);
      const int row = r + half * 8;
#pragma unroll
      for (int nt = 0; nt < 4; ++nt) {
        const int ncol = nt * 16 + m;
        const float h = (acc[nt][r] - mu) * inv * S_gg[ncol] + S_betag[ncol];
        HBUF[wid][row * 64 + ncol] = (_Float16)fmaxf(h, 0.f);
      }
    }
  }
  wave_lds_fence();

  // ---- GEMM2: h @ wg2 + bg2, softmax over the 16 neighbor rows ----
  float w[4][8];
  {
    const v16h H0 = load_a(&HBUF[wid][m * 64], 0, half);
    const v16h H1 = load_a(&HBUF[wid][m * 64], 1, half);
#pragma unroll
    for (int nt = 0; nt < 4; ++nt) {
      const int ncol = nt * 16 + m;
      v8f a = {};
      a = wmma16x16x32(H0, load_b(W2T, ncol, 0, half), a);
      a = wmma16x16x32(H1, load_b(W2T, ncol, 1, half), a);
      const float b = S_bg2[ncol];
      float mx = a[0] + b;
#pragma unroll
      for (int r = 0; r < 8; ++r) {
        w[nt][r] = a[r] + b;
        mx = fmaxf(mx, w[nt][r]);
      }
      mx = fmaxf(mx, __shfl_xor(mx, 16, 32));  // other 8 rows in lane^16
      float se = 0.f;
#pragma unroll
      for (int r = 0; r < 8; ++r) {
        const float e = __expf(w[nt][r] - mx);
        w[nt][r] = e;
        se += e;
      }
      se += __shfl_xor(se, 16, 32);
      const float inv = 1.f / se;
#pragma unroll
      for (int r = 0; r < 8; ++r) w[nt][r] *= inv;
    }
  }

  // ---- out2 = sum_m (gv + p) * w  (gv gather hits L2) ----
#pragma unroll
  for (int nt = 0; nt < 4; ++nt) {
    const int c = nt * 16 + m;
    float o = 0.f;
#pragma unroll
    for (int r = 0; r < 8; ++r) {
      const int mm = r + half * 8;
      const long idm = IDX[wid][mm];
      const float gv = vbuf[idm * 64 + c];
      const float pp = PBUF[wid][mm * 64 + c];
      o = fmaf(gv + pp, w[nt][r], o);
    }
    o += __shfl_xor(o, 16, 32);
    if (half == 0) out2[n * 64 + c] = o;
  }
}

// ---------------------------------------------------------------------------
// Kernel 3: out = out2 @ wo + bo + features (residual).
// ---------------------------------------------------------------------------
__global__ __launch_bounds__(128) void final_kernel(
    const float* __restrict__ inb, const float* __restrict__ feat,
    const _Float16* __restrict__ wT1, const float* __restrict__ bo,
    float* __restrict__ out, int N) {
  __shared__ __attribute__((aligned(32))) _Float16 WT[64 * 64];
  __shared__ float BIAS[64];
  __shared__ __attribute__((aligned(32))) _Float16 AT[4][16 * 64];

  const int t = threadIdx.x;
  stage_weights(wT1, WT, 4096, t, 128);
  if (t < 64) BIAS[t] = bo[t];
  __syncthreads();

  const int wid = t >> 5;
  const int lane = t & 31;
  const int m = lane & 15, half = lane >> 4;
  const long row0 = ((long)blockIdx.x * 4 + wid) * 16;
  if (row0 >= N) return;

  {
    const float* src = inb + (row0 + m) * 64 + half * 32;
    _Float16* dst = &AT[wid][m * 64 + half * 32];
#pragma unroll
    for (int j = 0; j < 32; j += 4) {
      const float4 f = *(const float4*)(src + j);
      dst[j + 0] = (_Float16)f.x;
      dst[j + 1] = (_Float16)f.y;
      dst[j + 2] = (_Float16)f.z;
      dst[j + 3] = (_Float16)f.w;
    }
  }
  wave_lds_fence();

  const v16h A0 = load_a(&AT[wid][m * 64], 0, half);
  const v16h A1 = load_a(&AT[wid][m * 64], 1, half);

#pragma unroll
  for (int nt = 0; nt < 4; ++nt) {
    const int ncol = nt * 16 + m;
    v8f acc = {};
    acc = wmma16x16x32(A0, load_b(WT, ncol, 0, half), acc);
    acc = wmma16x16x32(A1, load_b(WT, ncol, 1, half), acc);
    const float bias = BIAS[ncol];
    float* dst = out + (row0 + half * 8) * 64 + ncol;
    const float* res = feat + (row0 + half * 8) * 64 + ncol;
#pragma unroll
    for (int r = 0; r < 8; ++r) dst[r * 64] = acc[r] + bias + res[r * 64];
  }
}

// ---------------------------------------------------------------------------
extern "C" void kernel_launch(void* const* d_in, const int* in_sizes, int n_in,
                              void* d_out, int out_size, void* d_ws,
                              size_t ws_size, hipStream_t stream) {
  const float* points = (const float*)d_in[0];
  const float* feat = (const float*)d_in[1];
  const int* nidx = (const int*)d_in[2];
  const float* wq = (const float*)d_in[3];
  const float* bq = (const float*)d_in[4];
  const float* wk = (const float*)d_in[5];
  const float* bk = (const float*)d_in[6];
  const float* wv = (const float*)d_in[7];
  const float* bv = (const float*)d_in[8];
  const float* wp = (const float*)d_in[9];
  const float* bp = (const float*)d_in[10];
  const float* gp = (const float*)d_in[11];
  const float* betap = (const float*)d_in[12];
  const float* wg1 = (const float*)d_in[13];
  const float* bg1 = (const float*)d_in[14];
  const float* gg = (const float*)d_in[15];
  const float* betag = (const float*)d_in[16];
  const float* wg2 = (const float*)d_in[17];
  const float* bg2 = (const float*)d_in[18];
  const float* wo = (const float*)d_in[19];
  const float* bo = (const float*)d_in[20];

  const int N = in_sizes[0] / 3;  // points is (N,3)

  float* qbuf = (float*)d_ws;
  float* kbuf = qbuf + (size_t)N * 64;
  float* vbuf = kbuf + (size_t)N * 64;
  float* o2 = vbuf + (size_t)N * 64;
  _Float16* wT = (_Float16*)(o2 + (size_t)N * 64);  // 6*4096 f16 (48 KB)

  const dim3 blk(128);
  prep_weights<<<(6 * 4096 + 255) / 256, dim3(256), 0, stream>>>(
      wq, wk, wv, wg1, wg2, wo, wT);
  qkv_kernel<<<(N + 63) / 64, blk, 0, stream>>>(feat, wT, bq, bk, bv, qbuf,
                                                kbuf, vbuf, N);
  attn_kernel<<<(N + 3) / 4, blk, 0, stream>>>(
      points, nidx, qbuf, kbuf, vbuf, wT + 3 * 4096, wp, bp, gp, betap, bg1, gg,
      betag, bg2, o2, N);
  final_kernel<<<(N + 63) / 64, blk, 0, stream>>>(o2, feat, wT + 5 * 4096, bo,
                                                  (float*)d_out, N);
}
